// ConvResidualBlock_84945863180686
// MI455X (gfx1250) — matrix operand order repaired and checked by reading, not compiled, not run on hardware
//
#include <hip/hip_runtime.h>
#include <hip/hip_bf16.h>
#include <math.h>

#define FDIM 128

typedef __bf16 bf16_t;
typedef __attribute__((ext_vector_type(16))) __bf16 v16bf;
typedef __attribute__((ext_vector_type(8)))  __bf16 v8bf;
typedef __attribute__((ext_vector_type(8)))  float  v8f;

typedef unsigned int u32x4 __attribute__((ext_vector_type(4)));
typedef int          s32x4 __attribute__((ext_vector_type(4)));
typedef int          s32x8 __attribute__((ext_vector_type(8)));

// TDM availability: device pass of clang-22/23 declares the builtin; host pass
// does not (falls through to the portable body). Arity differs per toolchain.
#if defined(__has_builtin)
# if __has_builtin(__builtin_amdgcn_tensor_load_to_lds)
#  define HAVE_TDM 1
# endif
#endif
#ifndef HAVE_TDM
# define HAVE_TDM 0
#endif
#if __has_include(<hip/amd_detail/amd_gfx1250_TDM.h>)
# define TDM_SIX_ARGS 1
#else
# define TDM_SIX_ARGS 0
#endif

// LDS row stride: 256B of bf16 payload + 16B TDM pad -> ds bank = (4r+c)%64,
// conflict-free across the 16 lanes that read a B fragment, 16B aligned.
#define ROWB      272
#define SW_BYTES  (4 * FDIM * ROWB)      // 4 weight matrices, 128 rows each
#define SA_BYTES  (16 * ROWB)            // one 16x128 A slab
#define SMEM_BYTES (SW_BYTES + SA_BYTES) // 143,616 B of 320KB WGP LDS

#if HAVE_TDM
// 2D tensor-DMA: rows x rowElems bf16, row-major, padded 4 dwords per 64 dwords.
__device__ __forceinline__ void tdm_load_2d(unsigned ldsOff, unsigned long long gaddr,
                                            unsigned rowElems, unsigned rows)
{
    u32x4 g0;
    g0[0] = 1u;                                            // count=1, no gather
    g0[1] = ldsOff;                                        // lds_addr (bytes)
    g0[2] = (unsigned)(gaddr & 0xffffffffull);             // global_addr lo
    g0[3] = (unsigned)((gaddr >> 32) & 0x1ffffffull)       // global_addr hi
          | (2u << 30);                                    // type=2 ("image")
    s32x8 g1;
    g1[0] = (int)(0x00010000u                              // data_size=1 (2B)
          | (1u << 20)                                     // pad_enable
          | (5u << 22)                                     // pad_interval: 64 dw
          | (3u << 25));                                   // pad_amount: 4 dw
    g1[1] = (int)((rowElems & 0xffffu) << 16);             // tensor_dim0[15:0]
    g1[2] = (int)((rows & 0xffffu) << 16);                 // tensor_dim1[15:0]
    g1[3] = (int)((rowElems & 0xffffu) << 16);             // tile_dim0
    g1[4] = (int)(rows & 0xffffu);                         // tile_dim1, tile_dim2=0
    g1[5] = (int)rowElems;                                 // tensor_dim0_stride lo32
    g1[6] = 0;
    g1[7] = 0;
    s32x4 z4 = {0, 0, 0, 0};
#if TDM_SIX_ARGS
    s32x8 z8 = {0, 0, 0, 0, 0, 0, 0, 0};
    __builtin_amdgcn_tensor_load_to_lds(g0, g1, z4, z4, z8, 0);
#else
    __builtin_amdgcn_tensor_load_to_lds(g0, g1, z4, z4, 0);
#endif
}
#endif

__device__ __forceinline__ v16bf load_frag16(const char* p, int hiOff) {
    v8bf lo = *(const v8bf*)p;
    v8bf hi = *(const v8bf*)(p + hiOff);
    return __builtin_shufflevector(lo, hi, 0,1,2,3,4,5,6,7,8,9,10,11,12,13,14,15);
}

__device__ __forceinline__ void atomicMaxFloat(float* addr, float val) {
    if (val >= 0.0f) atomicMax((int*)addr, __float_as_int(val));
    else             atomicMin((unsigned int*)addr, (unsigned int)__float_as_int(val));
}

// ---------------------------------------------------------------------------
// f32 -> bf16 pre-conversion (weights once, X for the first layer only)
// ---------------------------------------------------------------------------
__global__ __launch_bounds__(256)
void f32_to_bf16_kernel(const float* __restrict__ src, bf16_t* __restrict__ dst, int n)
{
    int i = blockIdx.x * blockDim.x + threadIdx.x;
    if (i < n) dst[i] = (bf16_t)src[i];
}

// ---------------------------------------------------------------------------
// Fused 4-way GEMM: Q/K/V/H(skip) = X[N,128] @ W{q,k,v,s}^T + b.
// TDM path: stage all 4 bf16 weight matrices in LDS once per block (amortized
// over mPerBlock row tiles), stage each 16x128 A slab via TDM, feed WMMA from
// LDS. Fallback: direct global bf16 loads.
// Wmats = 4 contiguous [128,128] bf16 matrices in q,k,v,s order.
// ---------------------------------------------------------------------------
__global__ __launch_bounds__(256)
void gemm_qkvs_kernel(const bf16_t* __restrict__ Xb, const bf16_t* __restrict__ Wmats,
                      const float* __restrict__ Bq, const float* __restrict__ Bk,
                      const float* __restrict__ Bv, const float* __restrict__ Bs,
                      float* __restrict__ Q, float* __restrict__ Ko,
                      float* __restrict__ V, float* __restrict__ H,
                      int nTilesM, int mPerBlock)
{
    const int wid  = threadIdx.x >> 5;       // 8 waves = 8 column tiles
    const int lane = threadIdx.x & 31;
    const int h    = lane >> 4;              // half-wave 0/1
    const int r    = lane & 15;
    const int bcol = wid * 16 + r;           // output column (B-matrix N index)

    const float biasq = Bq[bcol], biask = Bk[bcol], biasv = Bv[bcol], biass = Bs[bcol];

#if HAVE_TDM
    extern __shared__ char smem[];
    char* sW = smem;                         // 4 * 128 * ROWB
    char* sA = smem + SW_BYTES;              // 16 * ROWB

    if (wid == 0) {
        #pragma unroll
        for (int m = 0; m < 4; ++m)
            tdm_load_2d((unsigned)(size_t)(sW + m * FDIM * ROWB),
                        (unsigned long long)(size_t)(Wmats + m * FDIM * FDIM),
                        FDIM, FDIM);
    }
#endif

    for (int it = 0; it < mPerBlock; ++it) {
        const int mtile = blockIdx.x * mPerBlock + it;
        if (mtile >= nTilesM) break;

#if HAVE_TDM
        if (wid == 0) {
            tdm_load_2d((unsigned)(size_t)sA,
                        (unsigned long long)(size_t)(Xb + (size_t)mtile * 16 * FDIM),
                        FDIM, 16);
            __builtin_amdgcn_s_wait_tensorcnt(0);   // W loads (first iter) + A load
        }
        __syncthreads();
#endif

        v8f cq = {0,0,0,0,0,0,0,0};
        v8f ck = cq, cv = cq, cs = cq;

        #pragma unroll
        for (int k0 = 0; k0 < FDIM; k0 += 32) {
#if HAVE_TDM
            // A fragment: lane(h,r) elem e -> K = 16h+e
            const char* ap = sA + r * ROWB + (k0 + 16 * h) * 2;
            v16bf a = load_frag16(ap, 16);
            // B fragment: elem e -> K = k0+16h+e, row bcol of W (B = W^T)
            const char* bp = sW + bcol * ROWB + (k0 + 16 * h) * 2;
            v16bf fbq = load_frag16(bp + 0 * FDIM * ROWB, 16);
            v16bf fbk = load_frag16(bp + 1 * FDIM * ROWB, 16);
            v16bf fbv = load_frag16(bp + 2 * FDIM * ROWB, 16);
            v16bf fbs = load_frag16(bp + 3 * FDIM * ROWB, 16);
#else
            const int arow = mtile * 16 + r;
            const char* ap = (const char*)(Xb + (size_t)arow * FDIM + k0 + 16 * h);
            v16bf a = load_frag16(ap, 16);
            const char* bp = (const char*)(Wmats + (size_t)bcol * FDIM + k0 + 16 * h);
            v16bf fbq = load_frag16(bp + 0 * FDIM * FDIM * 2, 16);
            v16bf fbk = load_frag16(bp + 1 * FDIM * FDIM * 2, 16);
            v16bf fbv = load_frag16(bp + 2 * FDIM * FDIM * 2, 16);
            v16bf fbs = load_frag16(bp + 3 * FDIM * FDIM * 2, 16);
#endif
            cq = __builtin_amdgcn_wmma_f32_16x16x32_bf16(false, a, false, fbq, (short)0, cq, false, false);
            ck = __builtin_amdgcn_wmma_f32_16x16x32_bf16(false, a, false, fbk, (short)0, ck, false, false);
            cv = __builtin_amdgcn_wmma_f32_16x16x32_bf16(false, a, false, fbv, (short)0, cv, false, false);
            cs = __builtin_amdgcn_wmma_f32_16x16x32_bf16(false, a, false, fbs, (short)0, cs, false, false);
        }

        // C/D layout: VGPR v, lanes 0-15 -> M=v ; lanes 16-31 -> M=v+8 ; N=lane&15
        #pragma unroll
        for (int v = 0; v < 8; ++v) {
            const int orow = mtile * 16 + v + 8 * h;
            const size_t o = (size_t)orow * FDIM + bcol;
            Q[o]  = cq[v] + biasq;
            Ko[o] = ck[v] + biask;
            V[o]  = cv[v] + biasv;
            H[o]  = cs[v] + biass;
        }
#if HAVE_TDM
        __syncthreads();   // all waves done reading sA before next TDM overwrite
#endif
    }
}

// ---------------------------------------------------------------------------
__global__ __launch_bounds__(256)
void init_kernel(float* __restrict__ M, float* __restrict__ Dn,
                 float* __restrict__ St, int n)
{
    int i = blockIdx.x * blockDim.x + threadIdx.x;
    if (i < n) { M[i] = __int_as_float(0xff800000u); Dn[i] = 0.0f; }
    if (i < 256) St[i] = 0.0f;
}

// alpha[e] = dot(q[dst], k[src]) / sqrt(128); segment max via float atomic-max.
__global__ __launch_bounds__(256)
void edge_alpha_kernel(const long long* __restrict__ edges,
                       const float* __restrict__ Q, const float* __restrict__ K,
                       float* __restrict__ Al, float* __restrict__ M, int E)
{
    const int e    = (blockIdx.x * blockDim.x + threadIdx.x) >> 5;
    const int lane = threadIdx.x & 31;
    if (e >= E) return;
    const long long s = edges[e];
    const long long d = edges[E + e];

    const float4 q4 = ((const float4*)(Q + (size_t)d * FDIM))[lane];
    const float4 k4 = ((const float4*)(K + (size_t)s * FDIM))[lane];
    float sum = q4.x * k4.x + q4.y * k4.y + q4.z * k4.z + q4.w * k4.w;
    sum += __shfl_xor(sum, 16);
    sum += __shfl_xor(sum, 8);
    sum += __shfl_xor(sum, 4);
    sum += __shfl_xor(sum, 2);
    sum += __shfl_xor(sum, 1);
    if (lane == 0) {
        const float al = sum * 0.08838834764831845f;  // 1/sqrt(128)
        Al[e] = al;
        atomicMaxFloat(M + d, al);
    }
}

// ea = exp(alpha - m[dst]); denom[dst] += ea.
__global__ __launch_bounds__(256)
void edge_exp_kernel(const long long* __restrict__ edges,
                     float* __restrict__ Al, const float* __restrict__ M,
                     float* __restrict__ Dn, int E)
{
    int e = blockIdx.x * blockDim.x + threadIdx.x;
    if (e >= E) return;
    const long long d = edges[E + e];
    const float ea = __expf(Al[e] - M[d]);
    Al[e] = ea;
    unsafeAtomicAdd(Dn + d, ea);
}

// a = ea / (denom[dst] + eps); H[dst,:] += a * V[src,:]; accumulate alpha out.
__global__ __launch_bounds__(256)
void edge_scatter_kernel(const long long* __restrict__ edges,
                         const float* __restrict__ Al, const float* __restrict__ Dn,
                         const float* __restrict__ V, float* __restrict__ H,
                         float* __restrict__ As, int E, int accumulate)
{
    const int e    = (blockIdx.x * blockDim.x + threadIdx.x) >> 5;
    const int lane = threadIdx.x & 31;
    if (e >= E) return;
    const long long s = edges[e];
    const long long d = edges[E + e];
    const float a = Al[e] / (Dn[d] + 1e-16f);
    if (lane == 0) {
        if (accumulate) As[e] += a; else As[e] = a;
    }
    const float4 v4 = ((const float4*)(V + (size_t)s * FDIM))[lane];
    float* hp = H + (size_t)d * FDIM + lane * 4;
    unsafeAtomicAdd(hp + 0, a * v4.x);
    unsafeAtomicAdd(hp + 1, a * v4.y);
    unsafeAtomicAdd(hp + 2, a * v4.z);
    unsafeAtomicAdd(hp + 3, a * v4.w);
}

// BN statistics: register accumulation per column, 2 atomics/thread/block.
__global__ __launch_bounds__(128)
void bn_stats_kernel(const float* __restrict__ H, float* __restrict__ St, int n)
{
    const int c = threadIdx.x;
    float s = 0.0f, s2 = 0.0f;
    for (int r = blockIdx.x; r < n; r += gridDim.x) {
        const float v = H[(size_t)r * FDIM + c];
        s += v; s2 += v * v;
    }
    unsafeAtomicAdd(St + c, s);
    unsafeAtomicAdd(St + FDIM + c, s2);
}

__global__ __launch_bounds__(128)
void bn_finalize_kernel(float* __restrict__ St, int n)
{
    const int c = threadIdx.x;
    const float inv_n = 1.0f / (float)n;
    const float mu  = St[c] * inv_n;
    const float var = fmaxf(St[FDIM + c] * inv_n - mu * mu, 0.0f);
    St[c]        = mu;
    St[FDIM + c] = rsqrtf(var + 1e-5f);
}

// mode 0: out = relu(bn(h)), also emit bf16 copy (next layer's GEMM input)
// mode 1: out = relu(bn(h) + x0)  (final output)
__global__ __launch_bounds__(256)
void bn_apply_kernel(const float* __restrict__ H, const float* __restrict__ St,
                     const float* __restrict__ G, const float* __restrict__ Be,
                     const float* __restrict__ X0, float* __restrict__ Out,
                     bf16_t* __restrict__ XbOut, int nd, int mode)
{
    int i = blockIdx.x * blockDim.x + threadIdx.x;
    if (i >= nd) return;
    const int c = i & (FDIM - 1);
    float v = (H[i] - St[c]) * St[FDIM + c] * G[c] + Be[c];
    if (mode == 1) v += X0[i];
    v = fmaxf(v, 0.0f);
    Out[i] = v;
    if (mode == 0) XbOut[i] = (bf16_t)v;
}

// Outputs 2 and 3 of the tuple: edges (as float) and a1+a2, packed after x.
__global__ __launch_bounds__(256)
void tail_kernel(const long long* __restrict__ edges, const float* __restrict__ As,
                 float* __restrict__ Out, int nd, int E, int out_size)
{
    int i = blockIdx.x * blockDim.x + threadIdx.x;
    if (i < 2 * E) {
        int o = nd + i;
        if (o < out_size) Out[o] = (float)edges[i];
    } else if (i < 3 * E) {
        int e = i - 2 * E;
        int o = nd + 2 * E + e;
        if (o < out_size) Out[o] = As[e];
    }
}

// ---------------------------------------------------------------------------
extern "C" void kernel_launch(void* const* d_in, const int* in_sizes, int n_in,
                              void* d_out, int out_size, void* d_ws, size_t ws_size,
                              hipStream_t stream)
{
    const float*     x0    = (const float*)d_in[0];
    const long long* edges = (const long long*)d_in[1];
    const int ND = in_sizes[0];
    const int N  = ND / FDIM;
    const int E  = in_sizes[1] / 2;

    const float *W[2][4], *B[2][4];
    for (int l = 0; l < 2; ++l)
        for (int j = 0; j < 4; ++j) {         // order: q,k,v,s
            W[l][j] = (const float*)d_in[2 + l * 8 + j * 2];
            B[l][j] = (const float*)d_in[2 + l * 8 + j * 2 + 1];
        }
    const float* g1  = (const float*)d_in[18];
    const float* be1 = (const float*)d_in[19];
    const float* g2  = (const float*)d_in[20];
    const float* be2 = (const float*)d_in[21];

    // Workspace: 5*ND + 2N + 2E + 256 floats, then bf16 region (X, 8 weights)
    float* ws = (float*)d_ws;
    size_t off = 0;
    float* Q  = ws + off; off += (size_t)ND;
    float* K  = ws + off; off += (size_t)ND;
    float* V  = ws + off; off += (size_t)ND;
    float* H  = ws + off; off += (size_t)ND;
    float* X1 = ws + off; off += (size_t)ND;
    float* M  = ws + off; off += (size_t)N;
    float* Dn = ws + off; off += (size_t)N;
    float* Al = ws + off; off += (size_t)E;
    float* As = ws + off; off += (size_t)E;
    float* St = ws + off; off += 256;
    bf16_t* Xb  = (bf16_t*)(ws + off); off += (size_t)(ND + 1) / 2;
    bf16_t* Wb8 = (bf16_t*)(ws + off); off += (size_t)(8 * FDIM * FDIM) / 2;
    (void)ws_size; (void)n_in;

    float* outx = (float*)d_out;

    const int nTilesM    = N / 16;              // 50000/16 = 3125 exactly
    const int MT         = 5;                   // row tiles per GEMM block
    const int gemmBlocks = (nTilesM + MT - 1) / MT;
    const int waveBlocks = (E + 7) / 8;         // one wave per edge
    const int edgeBlocks = (E + 255) / 256;
    const int ndBlocks   = (ND + 255) / 256;
    const int initBlocks = (N + 255) / 256;
    const int wElems     = FDIM * FDIM;

    // One-time weight conversion to bf16 (reused 3125x per layer by the GEMM)
    for (int l = 0; l < 2; ++l)
        for (int j = 0; j < 4; ++j)
            f32_to_bf16_kernel<<<(wElems + 255) / 256, 256, 0, stream>>>(
                W[l][j], Wb8 + (size_t)(l * 4 + j) * wElems, wElems);

    for (int l = 0; l < 2; ++l) {
        float* bnOut    = (l == 0) ? X1 : outx;
        const float* g  = (l == 0) ? g1 : g2;
        const float* be = (l == 0) ? be1 : be2;

        if (l == 0)   // layer-2 Xb is produced fused inside bn_apply(mode 0)
            f32_to_bf16_kernel<<<ndBlocks, 256, 0, stream>>>(x0, Xb, ND);

        init_kernel<<<initBlocks, 256, 0, stream>>>(M, Dn, St, N);
        gemm_qkvs_kernel<<<gemmBlocks, 256, SMEM_BYTES, stream>>>(
            Xb, Wb8 + (size_t)l * 4 * wElems,
            B[l][0], B[l][1], B[l][2], B[l][3],
            Q, K, V, H, nTilesM, MT);
        edge_alpha_kernel<<<waveBlocks, 256, 0, stream>>>(edges, Q, K, Al, M, E);
        edge_exp_kernel<<<edgeBlocks, 256, 0, stream>>>(edges, Al, M, Dn, E);
        edge_scatter_kernel<<<waveBlocks, 256, 0, stream>>>(edges, Al, Dn, V, H, As, E, l);
        bn_stats_kernel<<<512, 128, 0, stream>>>(H, St, N);
        bn_finalize_kernel<<<1, 128, 0, stream>>>(St, N);
        bn_apply_kernel<<<ndBlocks, 256, 0, stream>>>(H, St, g, be, x0, bnOut, Xb, ND, l);
    }
    tail_kernel<<<(3 * E + 255) / 256, 256, 0, stream>>>(edges, As, outx, ND, E, out_size);
}
